// SigmoidFlow2D_81527069212731
// MI455X (gfx1250) — matrix operand order repaired and checked by reading, not compiled, not run on hardware
//
#include <hip/hip_runtime.h>
#include <hip/hip_bf16.h>
#include <math.h>

#define D 128
#define NMID 6
#define STR 136            // LDS activation row stride in halves (272B, conflict-free A loads)
#define TILE_HALVES 512    // one 32x16 B fragment tile: 32 lanes * 16 halves
#define MAT_HALVES (D*D)   // 16384 halves (32KB) per packed matrix

typedef __attribute__((ext_vector_type(16))) _Float16 v16h;
typedef __attribute__((ext_vector_type(8)))  _Float16 v8h;
typedef __attribute__((ext_vector_type(8)))  float    v8f;

union Frag16 { v16h v; uint4 q[2]; };

__device__ __forceinline__ float fast_sigmoid(float v) {
    return __builtin_amdgcn_rcpf(1.0f + __expf(-v));   // v_exp_f32 + v_add + v_rcp_f32
}

// ---------------------------------------------------------------------------
// Prep: row-softmax (optionally scaled by exp(log_a)) of every weight matrix,
// written in WMMA B-fragment order:  element (k,j) -> tile (kt=k/32, jt=j/16),
// lane=(j&15)+16*((k>>4)&1), half=k&15  => each lane's fragment is 32B contig.
// ---------------------------------------------------------------------------
__global__ __launch_bounds__(128) void prep_weights(
    const float* __restrict__ w0,  const float* __restrict__ uM,
    const float* __restrict__ wM,  const float* __restrict__ laM,
    const float* __restrict__ uL,  const float* __restrict__ laL,
    _Float16* __restrict__ wsPack, float* __restrict__ wsCuL)
{
    __shared__ float red[128];
    const int tid = threadIdx.x;
    const int job = blockIdx.x;

    const float* src;
    _Float16* dst = nullptr;
    float scale = 1.0f;
    int j = 0;
    bool isCuL = false;

    if (job < 128) {                       // W0T = softmax(w0 rows), transposed
        j = job;  src = w0 + (size_t)j * D;
        dst = wsPack;
    } else if (job < 128 + NMID * 256) {   // middle: B1 = a*softmax(u), B2 = softmax(w)
        int idx   = job - 128;
        int l     = idx >> 8;
        int rest  = idx & 255;
        int which = rest >> 7;             // 0 = u -> B1, 1 = w -> B2
        j = rest & 127;
        src = (which ? wM : uM) + ((size_t)l * D + j) * D;
        if (!which) scale = __expf(laM[l * D + j]);
        dst = wsPack + MAT_HALVES + (size_t)l * 2 * MAT_HALVES + (which ? MAT_HALVES : 0);
    } else {                               // cuL = exp(laL)*softmax(uL row)
        isCuL = true;
        src = uL;
        scale = __expf(laL[0]);
    }

    float v = src[tid];
    red[tid] = v; __syncthreads();
    #pragma unroll
    for (int s = 64; s > 0; s >>= 1) {
        if (tid < s) red[tid] = fmaxf(red[tid], red[tid + s]);
        __syncthreads();
    }
    float mx = red[0]; __syncthreads();
    float e = __expf(v - mx);
    red[tid] = e; __syncthreads();
    #pragma unroll
    for (int s = 64; s > 0; s >>= 1) {
        if (tid < s) red[tid] += red[tid + s];
        __syncthreads();
    }
    float val = scale * e / red[0];

    if (isCuL) { wsCuL[tid] = val; return; }

    const int k    = tid;
    const int kt   = k >> 5;
    const int jt   = j >> 4;
    const int lane = (j & 15) + (((k >> 4) & 1) << 4);
    const int half = k & 15;
    dst[(size_t)(kt * 8 + jt) * TILE_HALVES + lane * 16 + half] = (_Float16)val;
}

// ---------------------------------------------------------------------------
// Async global->LDS staging of one packed 32KB weight matrix.
// 256 threads x 8 x b128 = 32KB; tracked with ASYNCcnt (8 outstanding / wave).
// ---------------------------------------------------------------------------
__device__ __forceinline__ void prefetch_B(const _Float16* __restrict__ gsrc,
                                           _Float16* ldsDst, int tid)
{
    const uint32_t lbase = (uint32_t)(uintptr_t)ldsDst;   // addr[31:0] = LDS offset
    #pragma unroll
    for (int i = 0; i < 8; ++i) {
        const uint32_t off = (uint32_t)(tid * 16 + i * 4096);
        asm volatile("global_load_async_to_lds_b128 %0, %1, %2"
                     :: "v"(lbase + off), "v"(off), "s"((const void*)gsrc)
                     : "memory");
    }
}
__device__ __forceinline__ void wait_async_le8() {
    asm volatile("s_wait_asynccnt 8" ::: "memory");
}
__device__ __forceinline__ void wait_async_0() {
    asm volatile("s_wait_asynccnt 0" ::: "memory");
}

// ---------------------------------------------------------------------------
// One fused 128x128 matmul stage, everything in LDS.
// 8 waves; wave = m-tile; 8 j-tiles; K=128 as 4 x WMMA(16x16x32 f16, f32 acc).
// SIG is compile-time: bias + sigmoid fused in f32 epilogue (biasLds in LDS).
// ---------------------------------------------------------------------------
template <bool SIG>
__device__ __forceinline__ void wmma_stage(
    const _Float16* __restrict__ src, _Float16* __restrict__ dst,
    const _Float16* __restrict__ Bpack, const float* __restrict__ biasLds,
    int lane, int wave)
{
    const int m  = wave * 16 + (lane & 15);
    const int kg = lane >> 4;

    Frag16 A[4];
    #pragma unroll
    for (int kt = 0; kt < 4; ++kt) {
        const int kb = kt * 32 + kg * 16;                // halves 0..15: K=kb..kb+15
        A[kt].q[0] = *(const uint4*)(src + m * STR + kb);
        A[kt].q[1] = *(const uint4*)(src + m * STR + kb + 8); // halves 8..15
    }

    #pragma unroll
    for (int jt = 0; jt < 8; ++jt) {
        v8f acc = {0.f, 0.f, 0.f, 0.f, 0.f, 0.f, 0.f, 0.f};
        #pragma unroll
        for (int kt = 0; kt < 4; ++kt) {
            Frag16 B;
            const _Float16* bp = Bpack + (kt * 8 + jt) * TILE_HALVES + lane * 16;
            B.q[0] = *(const uint4*)(bp);
            B.q[1] = *(const uint4*)(bp + 8);
            acc = __builtin_amdgcn_wmma_f32_16x16x32_f16(
                false, A[kt].v, false, B.v, (short)0, acc, false, false);
        }
        const int jc = jt * 16 + (lane & 15);
        float bj = 0.0f;
        if (SIG) bj = biasLds[jc];                       // LDS, hoisted, no branch
        #pragma unroll
        for (int r = 0; r < 8; ++r) {
            const int row = wave * 16 + r + 8 * kg;      // C/D: M = r + 8*(lane>=16)
            float v = acc[r];
            if (SIG) v = fast_sigmoid(v + bj);
            dst[row * STR + jc] = (_Float16)v;
        }
    }
}

// ---------------------------------------------------------------------------
// Fused flow: MLP theta -> first layer -> 6 middle layers -> final dot.
// One block = 128 rows; activations ping-pong between two LDS buffers;
// weight matrices double-buffered in LDS via async global->LDS copies.
// ---------------------------------------------------------------------------
__global__ __launch_bounds__(256) void flow_main(
    const float* __restrict__ x,   const float* __restrict__ fy0,
    const float* __restrict__ u0,  const float* __restrict__ la0,
    const float* __restrict__ b0,  const float* __restrict__ bM,
    const float* __restrict__ bL,
    const float* __restrict__ W1,  const float* __restrict__ b1,
    const float* __restrict__ W2,  const float* __restrict__ b2,
    const float* __restrict__ W3,  const float* __restrict__ b3,
    const _Float16* __restrict__ wsPack, const float* __restrict__ wsCuL,
    float* __restrict__ out, int Ntot)
{
    __shared__ __align__(16) _Float16 bufA[128 * STR];
    __shared__ __align__(16) _Float16 bufB[128 * STR];
    __shared__ __align__(16) _Float16 sB[2 * MAT_HALVES];   // double-buffered weights
    __shared__ float sW2[2500];
    __shared__ float sW1[50], sb1[50], sb2[50], sW3[50];
    __shared__ float sC0[128], sC1[128], sBb[128], sCuL[128];
    __shared__ float sBiasM[NMID * 128];                    // middle-layer biases

    const int tid  = threadIdx.x;
    const int rowg = blockIdx.x * 128 + (tid & 127);

    // global pointer of stage s's packed weight matrix (s = 0..12)
    auto stagePtr = [&](int s) -> const _Float16* {
        if (s == 0) return wsPack;
        const int l = (s - 1) >> 1;
        return wsPack + MAT_HALVES + (size_t)l * 2 * MAT_HALVES + ((s & 1) ? 0 : MAT_HALVES);
    };

    // kick off async staging of stage 0 weights immediately (buffer 0)
    prefetch_B(stagePtr(0), sB, tid);

    for (int i = tid; i < 2500; i += 256) sW2[i] = W2[i];
    for (int i = tid; i < NMID * 128; i += 256) sBiasM[i] = bM[i];
    if (tid < 50) { sW1[tid] = W1[tid]; sb1[tid] = b1[tid]; sb2[tid] = b2[tid]; sW3[tid] = W3[tid]; }
    if (tid < 128) {
        sCuL[tid] = wsCuL[tid];
        const int j = tid;
        float a0 = u0[2 * j], a1 = u0[2 * j + 1];
        float mx = fmaxf(a0, a1);
        float e0 = __expf(a0 - mx), e1 = __expf(a1 - mx);
        float inv = __expf(la0[j]) / (e0 + e1);
        sC0[j] = e0 * inv;
        sC1[j] = e1 * inv;
        sBb[j] = b0[j];
    }
    __syncthreads();

    // ---- phase 0: per-row MLP theta + first-layer inner (in=2), in VALU ----
    if (tid < 128) {
        const bool ok = rowg < Ntot;
        const float f = ok ? fy0[rowg] : 0.0f;
        float h1[50];
        #pragma unroll
        for (int i = 0; i < 50; ++i) h1[i] = fmaxf(f * sW1[i] + sb1[i], 0.0f);
        float theta = b3[0];
        for (int i = 0; i < 50; ++i) {          // dynamic outer, unrolled inner
            float acc = sb2[i];
            #pragma unroll
            for (int k = 0; k < 50; ++k) acc += h1[k] * sW2[i * 50 + k];
            theta += sW3[i] * fmaxf(acc, 0.0f);
        }
        const float xr = ok ? x[rowg] : 0.0f;
        for (int j = 0; j < 128; ++j) {
            float inner = xr * sC0[j] + theta * sC1[j] + sBb[j];
            bufA[tid * STR + j] = (_Float16)fast_sigmoid(inner);
        }
    }

    const int lane = tid & 31;
    const int wave = tid >> 5;

    // ---- stage 0: mono = s @ softmax(w0).T  (weights in sB buf0) ----
    prefetch_B(stagePtr(1), sB + MAT_HALVES, tid);   // B1 of layer 0 -> buf1
    wait_async_le8();                                // stage-0 group complete
    __syncthreads();                                 // + covers phase 0 writes
    wmma_stage<false>(bufA, bufB, sB, nullptr, lane, wave);
    __syncthreads();

    const _Float16* cur = bufB;
    _Float16*       nxt = bufA;

    // ---- 6 middle layers: sigmoid(h @ B1 + b) @ B2 ----
    for (int l = 0; l < NMID; ++l) {
        // B1 stage (odd stage -> buf1); prefetch B2 (even -> buf0)
        prefetch_B(stagePtr(2 + 2 * l), sB, tid);
        wait_async_le8();
        __syncthreads();
        wmma_stage<true>(cur, nxt, sB + MAT_HALVES, sBiasM + l * D, lane, wave);
        __syncthreads();
        { const _Float16* t = cur; cur = nxt; nxt = (_Float16*)t; }

        // B2 stage (even stage -> buf0); prefetch next B1 (odd -> buf1)
        if (l + 1 < NMID) {
            prefetch_B(stagePtr(3 + 2 * l), sB + MAT_HALVES, tid);
            wait_async_le8();
        } else {
            wait_async_0();
        }
        __syncthreads();
        wmma_stage<false>(cur, nxt, sB, nullptr, lane, wave);
        __syncthreads();
        { const _Float16* t = cur; cur = nxt; nxt = (_Float16*)t; }
    }

    // ---- final layer: out = sigmoid(dot(h, cuL) + bL)   (softmax(wL)==1) ----
    if (tid < 128 && rowg < Ntot) {
        const _Float16* rowp = cur + tid * STR;
        float acc = bL[0];
        #pragma unroll
        for (int c = 0; c < 16; ++c) {               // 16 x ds_load_b128
            v8h chunk = *(const v8h*)(rowp + c * 8);
            #pragma unroll
            for (int i = 0; i < 8; ++i)
                acc += (float)chunk[i] * sCuL[c * 8 + i];
        }
        out[rowg] = fast_sigmoid(acc);
    }
}

// ---------------------------------------------------------------------------
extern "C" void kernel_launch(void* const* d_in, const int* in_sizes, int n_in,
                              void* d_out, int out_size, void* d_ws, size_t ws_size,
                              hipStream_t stream)
{
    const float* x   = (const float*)d_in[0];
    const float* fy0 = (const float*)d_in[1];
    const float* u0  = (const float*)d_in[2];
    const float* w0  = (const float*)d_in[3];
    const float* la0 = (const float*)d_in[4];
    const float* b0  = (const float*)d_in[5];
    const float* uM  = (const float*)d_in[6];
    const float* wM  = (const float*)d_in[7];
    const float* laM = (const float*)d_in[8];
    const float* bM  = (const float*)d_in[9];
    const float* uL  = (const float*)d_in[10];
    // d_in[11] = wL: softmax over a 1-wide row == 1.0, mathematically unused
    const float* laL = (const float*)d_in[12];
    const float* bL  = (const float*)d_in[13];
    const float* W1  = (const float*)d_in[14];
    const float* b1  = (const float*)d_in[15];
    const float* W2  = (const float*)d_in[16];
    const float* b2  = (const float*)d_in[17];
    const float* W3  = (const float*)d_in[18];
    const float* b3  = (const float*)d_in[19];

    const int Ntot = in_sizes[0];

    // workspace: [W0T | 6 x (B1,B2)] packed f16, then cuL f32
    _Float16* wsPack = (_Float16*)d_ws;
    const size_t packHalves = (size_t)MAT_HALVES * (1 + 2 * NMID); // 212992 halves
    float* wsCuL = (float*)((char*)d_ws + packHalves * sizeof(_Float16));

    // 1) weight prep: one block per softmaxed row (+1 for uL)
    const int prepBlocks = 128 + NMID * 256 + 1;
    prep_weights<<<prepBlocks, 128, 0, stream>>>(w0, uM, wM, laM, uL, laL, wsPack, wsCuL);

    // 2) fused flow: one block per 128 rows
    const int blocks = (Ntot + 127) / 128;
    flow_main<<<blocks, 256, 0, stream>>>(x, fy0, u0, la0, b0, bM, bL,
                                          W1, b1, W2, b2, W3, b3,
                                          wsPack, wsCuL, (float*)d_out, Ntot);
}
